// ComponentAblationS4D_2929167696139
// MI455X (gfx1250) — matrix operand order repaired and checked by reading, not compiled, not run on hardware
//
#include <hip/hip_runtime.h>
#include <hip/hip_bf16.h>

// Problem constants (from reference)
static constexpr int B_  = 16;
static constexpr int L_  = 4096;
static constexpr int DIN = 8;
static constexpr int H_  = 256;
static constexpr int N2_ = 32;
static constexpr int NL_ = 4;

typedef __attribute__((ext_vector_type(16))) __bf16     v16bf;
typedef __attribute__((ext_vector_type(8)))  float      v8f;
typedef __attribute__((ext_vector_type(4)))  unsigned int v4u;

union FragBF { v16bf b; v4u q[2]; };

__device__ __forceinline__ unsigned short f2bf(float x) {
  unsigned int u = __float_as_uint(x);
  u += 0x7FFFu + ((u >> 16) & 1u);          // round-to-nearest-even
  return (unsigned short)(u >> 16);
}

// DPP8 selector for a lane-xor pattern within groups of 8
constexpr unsigned dpp8_sel_xor(int m) {
  unsigned r = 0;
  for (int i = 0; i < 8; ++i) r |= (unsigned)((i ^ m) & 7) << (3 * i);
  return r;
}

// c + (c shuffled by xor M in {1,2,4}) using DPP8 (VALU-rate cross-lane)
template <int M>
__device__ __forceinline__ float red_dpp8(float c) {
  int t = __builtin_amdgcn_mov_dpp8(__float_as_int(c), (int)dpp8_sel_xor(M));
  return c + __int_as_float(t);
}

// c + (c from lane^8) via v_permlane16_b32 (within each 16-lane row, VALU)
__device__ __forceinline__ float red_xor8(float c) {
  int ci = __float_as_int(c);
  int t = __builtin_amdgcn_permlane16(ci, ci, 0xFEDCBA98u, 0x76543210u,
                                      false, false);
  return c + __int_as_float(t);
}

// c + (c from lane^16) via v_permlanex16_b32 (swap 16-lane rows, VALU)
__device__ __forceinline__ float red_xor16(float c) {
  int ci = __float_as_int(c);
  int t = __builtin_amdgcn_permlanex16(ci, ci, 0x76543210u, 0xFEDCBA98u,
                                       false, false);
  return c + __int_as_float(t);
}

// ---------------------------------------------------------------------------
// Encoder: h[b,ch,l] = sum_i x[b,l,i] * enc_w[i,ch] + enc_b[ch]   (B,H,L) f32
// ---------------------------------------------------------------------------
__global__ __launch_bounds__(256) void enc_kernel(
    const float* __restrict__ x, const float* __restrict__ enc_w,
    const float* __restrict__ enc_b, float* __restrict__ h)
{
  size_t t = (size_t)blockIdx.x * 256 + threadIdx.x;   // (b, ch, l), l fastest
  int l  = (int)(t % L_);
  int ch = (int)((t / L_) % H_);
  int b  = (int)(t / ((size_t)L_ * H_));
  const float* xp = x + ((size_t)b * L_ + l) * DIN;
  float acc = enc_b[ch];
#pragma unroll
  for (int i = 0; i < DIN; ++i) acc = fmaf(xp[i], enc_w[i * H_ + ch], acc);
  h[t] = acc;
}

// ---------------------------------------------------------------------------
// out_w (NL,2H,H) f32 -> bf16, same layout (row-major, h contiguous)
// ---------------------------------------------------------------------------
__global__ __launch_bounds__(256) void wconv_kernel(
    const float* __restrict__ out_w, unsigned short* __restrict__ Wbf)
{
  size_t t = (size_t)blockIdx.x * 256 + threadIdx.x;
  Wbf[t] = f2bf(out_w[t]);
}

// ---------------------------------------------------------------------------
// S4D diagonal SSM scan: one wave per (b,ch), one lane per state n (wave32!).
// Per step: readlane input broadcast (VALU), complex state FMA, contribution,
// 5-stage reduction entirely on the VALU: 3x DPP8 + permlane16 + permlanex16.
// The hot loop issues ZERO ds-pipe instructions.
// Writes gelu(y) as bf16 directly in WMMA-B-fragment tile layout:
//   tile(b, hblk=ch/32, ltile=l/16) of 512 bf16:
//     offset = laneF*16 + t ; laneF = (l&15) + 16*((ch>>4)&1) ; t = ch&15
// ---------------------------------------------------------------------------
__global__ __launch_bounds__(256) void s4d_scan_kernel(
    const float* __restrict__ h,
    const float* __restrict__ log_dt,
    const float* __restrict__ C_re, const float* __restrict__ C_im,
    const float* __restrict__ log_A_real, const float* __restrict__ A_imag,
    const float* __restrict__ D_skip,
    unsigned short* __restrict__ Ybf, int li)
{
  int lane = threadIdx.x & 31;
  int wid  = threadIdx.x >> 5;
  int gw   = blockIdx.x * 8 + wid;
  int b    = gw >> 8;          // / H_
  int ch   = gw & (H_ - 1);

  // per-(ch,lane) discretized coefficients
  float dt  = __expf(log_dt[li * H_ + ch]);
  int   pi  = (li * H_ + ch) * N2_ + lane;
  float Are = -__expf(log_A_real[pi]);
  float Aim = A_imag[pi];
  float dre = Are * dt, dimm = Aim * dt;
  float em  = __expf(dre);
  float wre = em * __cosf(dimm);
  float wim = em * __sinf(dimm);
  float cr  = C_re[pi], ci = C_im[pi];
  float nr  = cr * (wre - 1.f) - ci * wim;          // C*(w-1)
  float ni  = cr * wim + ci * (wre - 1.f);
  float d2  = Are * Are + Aim * Aim;
  float inv = 2.0f / d2;                            // fold 2x into Ct
  float ctre = (nr * Are + ni * Aim) * inv;         // 2*Re/Im of C*(w-1)/A
  float ctim = (ni * Are - nr * Aim) * inv;
  float dsk  = D_skip[li * H_ + ch];

  const float* hrow = h + ((size_t)(b * H_ + ch)) * L_;
  unsigned short* base0 = Ybf
      + ((size_t)b * (H_ / 32) + (ch >> 5)) * (size_t)(L_ / 16) * 512
      + (size_t)((ch >> 4) & 1) * 256 + (ch & 15);

  float sre = 0.f, sim = 0.f;
  for (int l0 = 0; l0 < L_; l0 += 32) {
    float zv = hrow[l0 + lane];                     // coalesced 128B
    if (l0 + 128 < L_)
      __builtin_prefetch(hrow + l0 + 128 + lane, 0, 0);  // global_prefetch_b8
    float yk = 0.f;
#pragma unroll
    for (int j = 0; j < 32; ++j) {
      // broadcast step input via readlane (j is a compile-time constant)
      float z = __int_as_float(__builtin_amdgcn_readlane(__float_as_int(zv), j));
      float ns_im = fmaf(wre, sim, wim * sre);      // s = w*s + z
      float ns_re = fmaf(wre, sre, fmaf(-wim, sim, z));
      sre = ns_re; sim = ns_im;
      float c = fmaf(ctre, sre, -(ctim * sim));     // Re(Ct * s) (2x folded)
      c = red_dpp8<1>(c);                           // xor1  (DPP8, VALU)
      c = red_dpp8<2>(c);                           // xor2  (DPP8, VALU)
      c = red_dpp8<4>(c);                           // xor4  (DPP8, VALU)
      c = red_xor8(c);                              // xor8  (permlane16, VALU)
      c = red_xor16(c);                             // xor16 (permlanex16, VALU)
      float y = fmaf(dsk, z, c);                    // + D_skip * z
      if (j == lane) yk = y;                        // keep my timestep
    }
    float g = 0.5f * yk * (1.0f + erff(yk * 0.70710678118f)); // exact GELU
    int l = l0 + lane;
    base0[(size_t)(l >> 4) * 512 + (size_t)(l & 15) * 16] = f2bf(g);
  }
}

// ---------------------------------------------------------------------------
// 1x1 conv H->2H as bf16 WMMA GEMM + fused GLU.
// Per wave: one (b, o-tile) x FOUR l-tiles: A fragments (W rows, gate rows)
// are loaded once per K-step and reused by 4 B fragments -> 8 WMMA / 12 loads.
// A = Wbf rows (K contiguous), B = pre-tiled Ybf fragments (fully coalesced).
// ---------------------------------------------------------------------------
__global__ __launch_bounds__(256) void gemm_glu_kernel(
    const unsigned short* __restrict__ Wbf,
    const unsigned short* __restrict__ Ybf,
    const float* __restrict__ out_b,
    float* __restrict__ Z, int li)
{
  int lane = threadIdx.x & 31;
  int wid  = threadIdx.x >> 5;
  int wg   = blockIdx.x * 8 + wid;
  int lt4  = wg & (L_ / 64 - 1);        // 64 groups of 4 l-tiles
  int ot   = (wg >> 6) & 15;            // 16 o-tiles
  int b    = wg >> 10;
  int o0   = ot * 16;
  int rowA = lane & 15;
  int hi   = lane >> 4;                 // half-wave select
  int colOff = hi * 16;                 // A-frag K sub-offset (contiguous k = 16*hi + s)

  const unsigned short* aBase1 =
      Wbf + ((size_t)(li * 2 * H_ + o0 + rowA)) * H_ + colOff;
  const unsigned short* aBase2 = aBase1 + (size_t)H_ * H_;   // +256 rows (gate)
  const unsigned short* bBase =
      Ybf + ((size_t)b * (H_ / 32) * (L_ / 16) + lt4 * 4) * 512 + (size_t)lane * 16;

  v8f acc[2][4];
#pragma unroll
  for (int i = 0; i < 2; ++i)
#pragma unroll
    for (int t = 0; t < 4; ++t)
      acc[i][t] = (v8f){0.f, 0.f, 0.f, 0.f, 0.f, 0.f, 0.f, 0.f};

#pragma unroll
  for (int kb = 0; kb < H_ / 32; ++kb) {
    FragBF a1, a2;
    a1.q[0] = *(const v4u*)(aBase1 + kb * 32);        // K {0..7}|{16..23}
    a1.q[1] = *(const v4u*)(aBase1 + kb * 32 + 8);    // K {8..15}|{24..31}
    a2.q[0] = *(const v4u*)(aBase2 + kb * 32);
    a2.q[1] = *(const v4u*)(aBase2 + kb * 32 + 8);
    const unsigned short* bp = bBase + (size_t)kb * (L_ / 16) * 512;
#pragma unroll
    for (int t = 0; t < 4; ++t) {
      FragBF bb;
      bb.q[0] = *(const v4u*)(bp + t * 512);          // 32B contiguous/lane
      bb.q[1] = *(const v4u*)(bp + t * 512 + 8);
      acc[0][t] = __builtin_amdgcn_wmma_f32_16x16x32_bf16(
          false, a1.b, false, bb.b, (short)0, acc[0][t], false, false);
      acc[1][t] = __builtin_amdgcn_wmma_f32_16x16x32_bf16(
          false, a2.b, false, bb.b, (short)0, acc[1][t], false, false);
    }
  }

#pragma unroll
  for (int t = 0; t < 4; ++t) {
    int lcol = ((lt4 * 4 + t) << 4) + (lane & 15);
#pragma unroll
    for (int r = 0; r < 8; ++r) {
      int orow = o0 + r + hi * 8;                     // D-tile row mapping
      float a = acc[0][t][r] + out_b[li * 2 * H_ + orow];
      float g = acc[1][t][r] + out_b[li * 2 * H_ + H_ + orow];
      float z = a / (1.f + __expf(-g));               // GLU
      Z[((size_t)(b * H_ + orow)) * L_ + lcol] = z;
    }
  }
}

// ---------------------------------------------------------------------------
// Residual + channel LayerNorm, in place on h:  h = LN(Z + h)
// block (32 l, 8 ch-groups); each thread owns 32 channels at fixed l.
// ---------------------------------------------------------------------------
__global__ __launch_bounds__(256) void ln_res_kernel(
    const float* __restrict__ Z, float* __restrict__ h,
    const float* __restrict__ ln_g, const float* __restrict__ ln_b, int li)
{
  __shared__ float red[2][8][32];
  int tx = threadIdx.x, ty = threadIdx.y;
  int l = blockIdx.x * 32 + tx;
  int b = blockIdx.y;

  float vals[32];
  float s = 0.f, s2 = 0.f;
#pragma unroll
  for (int c = 0; c < 32; ++c) {
    int ch = ty * 32 + c;
    size_t idx = ((size_t)(b * H_ + ch)) * L_ + l;
    float v = Z[idx] + h[idx];
    vals[c] = v; s += v; s2 = fmaf(v, v, s2);
  }
  red[0][ty][tx] = s; red[1][ty][tx] = s2;
  __syncthreads();
  float ts = 0.f, ts2 = 0.f;
#pragma unroll
  for (int j = 0; j < 8; ++j) { ts += red[0][j][tx]; ts2 += red[1][j][tx]; }
  float mu = ts * (1.0f / H_);
  float var = ts2 * (1.0f / H_) - mu * mu;
  float rs = rsqrtf(var + 1e-5f);
#pragma unroll
  for (int c = 0; c < 32; ++c) {
    int ch = ty * 32 + c;
    size_t idx = ((size_t)(b * H_ + ch)) * L_ + l;
    h[idx] = (vals[c] - mu) * rs * ln_g[li * H_ + ch] + ln_b[li * H_ + ch];
  }
}

// ---------------------------------------------------------------------------
// Depthwise conv (K=5, same, no bias) + decoder GEMV (H->1), fused.
// ---------------------------------------------------------------------------
__global__ __launch_bounds__(256) void conv_dec_kernel(
    const float* __restrict__ h, const float* __restrict__ conv_w,
    const float* __restrict__ dec_w, const float* __restrict__ dec_b,
    float* __restrict__ out)
{
  __shared__ float red[8][32];
  int tx = threadIdx.x, ty = threadIdx.y;
  int l = blockIdx.x * 32 + tx;
  int b = blockIdx.y;

  float acc = 0.f;
#pragma unroll 4
  for (int c = 0; c < 32; ++c) {
    int ch = ty * 32 + c;
    const float* row = h + ((size_t)(b * H_ + ch)) * L_;
    float s = 0.f;
#pragma unroll
    for (int k = 0; k < 5; ++k) {
      int ll = l + k - 2;
      if (ll >= 0 && ll < L_) s = fmaf(conv_w[ch * 5 + k], row[ll], s);
    }
    acc = fmaf(s, dec_w[ch], acc);
  }
  red[ty][tx] = acc;
  __syncthreads();
  if (ty == 0) {
    float t = 0.f;
#pragma unroll
    for (int j = 0; j < 8; ++j) t += red[j][tx];
    out[(size_t)b * L_ + l] = t + dec_b[0];
  }
}

// ---------------------------------------------------------------------------
extern "C" void kernel_launch(void* const* d_in, const int* in_sizes, int n_in,
                              void* d_out, int out_size, void* d_ws, size_t ws_size,
                              hipStream_t stream)
{
  (void)in_sizes; (void)n_in; (void)out_size; (void)ws_size;
  const float* x          = (const float*)d_in[0];
  const float* enc_w      = (const float*)d_in[1];
  const float* enc_b      = (const float*)d_in[2];
  const float* log_dt     = (const float*)d_in[3];
  const float* C_re       = (const float*)d_in[4];
  const float* C_im       = (const float*)d_in[5];
  const float* log_A_real = (const float*)d_in[6];
  const float* A_imag     = (const float*)d_in[7];
  const float* D_skip     = (const float*)d_in[8];
  const float* out_w      = (const float*)d_in[9];
  const float* out_b      = (const float*)d_in[10];
  const float* ln_g       = (const float*)d_in[11];
  const float* ln_b       = (const float*)d_in[12];
  const float* conv_w     = (const float*)d_in[13];
  const float* dec_w      = (const float*)d_in[14];
  const float* dec_b      = (const float*)d_in[15];
  float* out = (float*)d_out;

  const size_t nBHL = (size_t)B_ * H_ * L_;          // 16.7M elements
  char* ws = (char*)d_ws;
  float* h            = (float*)ws;                          // 64 MiB
  float* Z            = (float*)(ws + nBHL * 4);             // 64 MiB
  unsigned short* Ybf = (unsigned short*)(ws + nBHL * 8);    // 32 MiB
  unsigned short* Wbf = (unsigned short*)(ws + nBHL * 10);   // 1 MiB

  enc_kernel<<<(unsigned)(nBHL / 256), 256, 0, stream>>>(x, enc_w, enc_b, h);
  wconv_kernel<<<(NL_ * 2 * H_ * H_) / 256, 256, 0, stream>>>(out_w, Wbf);

  for (int li = 0; li < NL_; ++li) {
    s4d_scan_kernel<<<B_ * H_ / 8, 256, 0, stream>>>(
        h, log_dt, C_re, C_im, log_A_real, A_imag, D_skip, Ybf, li);
    gemm_glu_kernel<<<(B_ * 16 * (L_ / 64)) / 8, 256, 0, stream>>>(
        Wbf, Ybf, out_b, Z, li);
    ln_res_kernel<<<dim3(L_ / 32, B_), dim3(32, 8), 0, stream>>>(
        Z, h, ln_g, ln_b, li);
  }

  conv_dec_kernel<<<dim3(L_ / 32, B_), dim3(32, 8), 0, stream>>>(
      h, conv_w, dec_w, dec_b, out);
}
